// DAGSelfAttention_49108656062901
// MI455X (gfx1250) — matrix operand order repaired and checked by reading, not compiled, not run on hardware
//
#include <hip/hip_runtime.h>

// ---------------------------------------------------------------------------
// Problem constants (from the reference)
// ---------------------------------------------------------------------------
static constexpr int Bb = 8;
static constexpr int Ll = 1024;
static constexpr int Dd = 256;
static constexpr int Hh = 8;
static constexpr int HD = 32;           // head dim
static constexpr int Mrows = Bb * Ll;   // 8192

// CDNA5 WMMA / vector types
typedef __bf16 v16bf __attribute__((ext_vector_type(16)));
typedef __bf16 v8bf  __attribute__((ext_vector_type(8)));
typedef float  v8f   __attribute__((ext_vector_type(8)));
typedef float  f32x4 __attribute__((ext_vector_type(4)));
typedef unsigned int u32x4 __attribute__((ext_vector_type(4)));
typedef int    i32x4 __attribute__((ext_vector_type(4)));
typedef int    i32x8 __attribute__((ext_vector_type(8)));

// Native hardware bf16 converts (v_cvt_pk_bf16_f32 path)
__device__ __forceinline__ __bf16 f2bf(float f) { return (__bf16)f; }
__device__ __forceinline__ float  bf2f(__bf16 b) { return (float)b; }

// ---------------------------------------------------------------------------
// Tensor Data Mover: 1-D tile of nElems 2-byte elements, global -> LDS.
// D# packed per CDNA5 ISA ch.8 (group0: count/lds_addr/global_addr/type,
// group1: data_size, tensor_dim0/1, tile_dim0/1, strides).
// ---------------------------------------------------------------------------
#if __has_builtin(__builtin_amdgcn_tensor_load_to_lds)
#define HAS_TDM 1
#else
#define HAS_TDM 0
#endif

#if __has_builtin(__builtin_amdgcn_s_wait_tensorcnt)
#define WAIT_TENSORCNT0() __builtin_amdgcn_s_wait_tensorcnt(0)
#else
#define WAIT_TENSORCNT0() asm volatile("s_wait_tensorcnt 0x0" ::: "memory")
#endif

#if HAS_TDM
__device__ __forceinline__ void tdm_load_1d(unsigned ldsAddr, const void* gptr,
                                            unsigned nElems) {
  unsigned long long ga = (unsigned long long)(__SIZE_TYPE__)gptr;
  u32x4 g0;
  g0[0] = 1u;                                        // count=1 (user D#)
  g0[1] = ldsAddr;                                   // lds_addr [63:32]
  g0[2] = (unsigned)(ga & 0xFFFFFFFFu);              // global_addr lo
  g0[3] = (unsigned)((ga >> 32) & 0x1FFFFFFu)        // global_addr hi
        | (2u << 30);                                // type = image
  i32x8 g1;
  g1[0] = (int)(1u << 16);                           // data_size=1 (2 bytes)
  g1[1] = (int)((nElems & 0xFFFFu) << 16);           // tensor_dim0 lo16
  g1[2] = (int)((nElems >> 16) | (1u << 16));        // dim0 hi16 | tensor_dim1=1
  g1[3] = (int)((nElems & 0xFFFFu) << 16);           // tile_dim0
  g1[4] = 1;                                         // tile_dim1=1
  g1[5] = (int)nElems;                               // tensor_dim0_stride lo32
  g1[6] = 0;
  g1[7] = 0;
  i32x4 z4 = {0, 0, 0, 0};
#if defined(__clang_major__) && (__clang_major__ >= 23)
  i32x8 z8 = {0, 0, 0, 0, 0, 0, 0, 0};
  __builtin_amdgcn_tensor_load_to_lds(g0, g1, z4, z4, z8, 0);
#else
  __builtin_amdgcn_tensor_load_to_lds(g0, g1, z4, z4, 0);
#endif
}
#endif  // HAS_TDM

// ---------------------------------------------------------------------------
// GEMM (wave per 16x32 tile): C = A @ W + bias, A f32 ([obs||act] concat on K),
// bf16 output in head-permuted [b,h,l,hd] layout.
// The concat boundary (Dd=256) is 32-aligned, so the A-source selection is
// UNIFORM per 32-wide K-step: pick a scalar base pointer once per iteration
// (s_cselect), never per lane.
// A-frag (16x32 bf16): lane m=lane&15 holds row m; element i -> K =
// (half<<4)+i == 16 contiguous elems.
// B-frag (32x16): lane holds col n=lane&15, element i -> K = (half<<4)+i.
// C tile: VGPR r -> row r / r+8 by lane half, col = lane&15.
// ---------------------------------------------------------------------------
__global__ void __launch_bounds__(32)
gemm_bf16out_kernel(const float* __restrict__ A1, const float* __restrict__ A2,
                    int K, const float* __restrict__ W,
                    const float* __restrict__ bias, __bf16* __restrict__ out) {
  const int lane  = threadIdx.x;
  const int laneN = lane & 15;
  const int half  = lane >> 4;
  const int m0 = blockIdx.x * 16;
  const int n0 = blockIdx.y * 32;
  const int rowA = m0 + laneN;

  v8f c0 = {}, c1 = {};
  for (int k0 = 0; k0 < K; k0 += 32) {
    // ---- uniform source pick for this K-step (entire 32-wide step is on
    //      one side of the 256-element concat boundary)
    const float* Abase = (k0 < Dd) ? A1 : A2;
    const int    kbase = (k0 < Dd) ? k0 : (k0 - Dd);
    const float* arow  = Abase + (size_t)rowA * Dd + kbase;
    // ---- A fragment: 16 contiguous floats (K = 16*half + i), converted to bf16
    f32x4 a0 = *(const f32x4*)(arow + (half << 4));
    f32x4 a1 = *(const f32x4*)(arow + (half << 4) + 4);
    f32x4 a2 = *(const f32x4*)(arow + (half << 4) + 8);
    f32x4 a3 = *(const f32x4*)(arow + (half << 4) + 12);
    v16bf a;
#pragma unroll
    for (int i = 0; i < 4; ++i) {
      a[i]      = f2bf(a0[i]);
      a[4 + i]  = f2bf(a1[i]);
      a[8 + i]  = f2bf(a2[i]);
      a[12 + i] = f2bf(a3[i]);
    }
    // ---- B fragments for the two 16-wide N tiles
    v16bf b0, b1;
#pragma unroll
    for (int i = 0; i < 16; ++i) {
      const int kb = k0 + (half << 4) + i;
      b0[i] = f2bf(W[(size_t)kb * Dd + n0 + laneN]);
      b1[i] = f2bf(W[(size_t)kb * Dd + n0 + 16 + laneN]);
    }
    if (k0 + 32 < K)  // next weight tile toward L2 (global_prefetch_b8)
      __builtin_prefetch(&W[(size_t)(k0 + 32) * Dd + n0 + laneN], 0, 0);
    c0 = __builtin_amdgcn_wmma_f32_16x16x32_bf16(false, a, false, b0,
                                                 (short)0, c0, false, false);
    c1 = __builtin_amdgcn_wmma_f32_16x16x32_bf16(false, a, false, b1,
                                                 (short)0, c1, false, false);
  }
  const float bv0 = bias[n0 + laneN];
  const float bv1 = bias[n0 + 16 + laneN];
#pragma unroll
  for (int r = 0; r < 8; ++r) {
    const int row = m0 + r + (half << 3);
    const int bI = row >> 10, l = row & (Ll - 1);
#pragma unroll
    for (int t = 0; t < 2; ++t) {
      const int col = n0 + t * 16 + laneN;
      const int hI = col >> 5, hd = col & (HD - 1);
      const float v = (t == 0 ? c0[r] + bv0 : c1[r] + bv1);
      out[((size_t)(bI * Hh + hI) * Ll + l) * HD + hd] = f2bf(v);
    }
  }
}

// ---------------------------------------------------------------------------
// Final projection: y(bf16) @ Wp + bp -> f32 d_out, plain [B*L, D] layout
// ---------------------------------------------------------------------------
__global__ void __launch_bounds__(32)
gemm_f32out_kernel(const __bf16* __restrict__ A, const float* __restrict__ W,
                   const float* __restrict__ bias, float* __restrict__ out) {
  const int lane  = threadIdx.x;
  const int laneN = lane & 15;
  const int half  = lane >> 4;
  const int m0 = blockIdx.x * 16;
  const int n0 = blockIdx.y * 32;
  const int rowA = m0 + laneN;

  v8f c0 = {}, c1 = {};
  for (int k0 = 0; k0 < Dd; k0 += 32) {
    const __bf16* ar = &A[(size_t)rowA * Dd + k0];
    v8bf ac0 = *(const v8bf*)(ar + (half << 4));
    v8bf ac1 = *(const v8bf*)(ar + (half << 4) + 8);
    v16bf a;
#pragma unroll
    for (int i = 0; i < 8; ++i) { a[i] = ac0[i]; a[8 + i] = ac1[i]; }
    v16bf b0, b1;
#pragma unroll
    for (int i = 0; i < 16; ++i) {
      const int kb = k0 + (half << 4) + i;
      b0[i] = f2bf(W[(size_t)kb * Dd + n0 + laneN]);
      b1[i] = f2bf(W[(size_t)kb * Dd + n0 + 16 + laneN]);
    }
    if (k0 + 32 < Dd)
      __builtin_prefetch(&W[(size_t)(k0 + 32) * Dd + n0 + laneN], 0, 0);
    c0 = __builtin_amdgcn_wmma_f32_16x16x32_bf16(false, a, false, b0,
                                                 (short)0, c0, false, false);
    c1 = __builtin_amdgcn_wmma_f32_16x16x32_bf16(false, a, false, b1,
                                                 (short)0, c1, false, false);
  }
#pragma unroll
  for (int r = 0; r < 8; ++r) {
    const int row = m0 + r + (half << 3);
    out[(size_t)row * Dd + n0 + laneN]      = c0[r] + bias[n0 + laneN];
    out[(size_t)row * Dd + n0 + 16 + laneN] = c1[r] + bias[n0 + 16 + laneN];
  }
}

// ---------------------------------------------------------------------------
// self_w[b,h,l] = <q[b,h,l,:], self_k[b,h,l,:]>
// ---------------------------------------------------------------------------
__global__ void selfw_kernel(const __bf16* __restrict__ q,
                             const __bf16* __restrict__ sk,
                             float* __restrict__ sw) {
  int i = blockIdx.x * blockDim.x + threadIdx.x;
  if (i >= Bb * Hh * Ll) return;
  float s = 0.f;
#pragma unroll
  for (int d = 0; d < HD; ++d)
    s += bf2f(q[(size_t)i * HD + d]) * bf2f(sk[(size_t)i * HD + d]);
  sw[i] = s;
}

// ---------------------------------------------------------------------------
// Flash-style attention, one wave per (b, h, 16-row q block).
// K/V tiles staged into LDS via the Tensor Data Mover; diagonal score replaced
// by self_w and its softmax weight routed to self_v.
// ---------------------------------------------------------------------------
__global__ void __launch_bounds__(32)
attn_kernel(const __bf16* __restrict__ qB, const __bf16* __restrict__ kB,
            const __bf16* __restrict__ vB, const __bf16* __restrict__ svB,
            const float* __restrict__ swB, const int* __restrict__ mask,
            __bf16* __restrict__ yB) {
  const int lane  = threadIdx.x;
  const int laneN = lane & 15;
  const int half  = lane >> 4;
  const int b  = blockIdx.z;
  const int h  = blockIdx.y;
  const int l0 = blockIdx.x * 16;
  const size_t headOff = (size_t)(b * Hh + h) * Ll;

  __shared__ __bf16 kLds[32 * HD];   // 32 key rows   (2 KB)
  __shared__ __bf16 vLds[32 * HD];   // 32 value rows (2 KB)
  __shared__ __bf16 pLds[16 * 32];   // P transpose staging (1 KB)

  // Q A-fragment (16 contiguous elems per lane, K = 16*half + i)
  v16bf aQ;
  {
    const __bf16* qrow = qB + (headOff + l0 + laneN) * HD;
    v8bf q0 = *(const v8bf*)(qrow + (half << 4));
    v8bf q1 = *(const v8bf*)(qrow + (half << 4) + 8);
#pragma unroll
    for (int i = 0; i < 8; ++i) { aQ[i] = q0[i]; aQ[8 + i] = q1[i]; }
  }

  float mrun[8], lrun[8], pdiag[8], sw[8];
#pragma unroll
  for (int r = 0; r < 8; ++r) {
    mrun[r] = -3.0e38f; lrun[r] = 0.f; pdiag[r] = 0.f;
    sw[r] = swB[headOff + l0 + r + (half << 3)];
  }
  v8f acc0 = {}, acc1 = {};
  const float scale = 0.1767766952966369f;  // 1/sqrt(32)

  for (int m0 = 0; m0 < Ll; m0 += 32) {
    // ---- stage 32 K rows + 32 V rows into LDS
#if HAS_TDM
    tdm_load_1d((unsigned)(__SIZE_TYPE__)&kLds[0],
                kB + (headOff + m0) * HD, 32u * HD);
    tdm_load_1d((unsigned)(__SIZE_TYPE__)&vLds[0],
                vB + (headOff + m0) * HD, 32u * HD);
    WAIT_TENSORCNT0();
#else
#pragma unroll
    for (int cch = 0; cch < 4; ++cch) {
      const int off = (cch * 32 + lane) * 8;
      *(v8bf*)(kLds + off) = *(const v8bf*)(kB + (headOff + m0) * HD + off);
      *(v8bf*)(vLds + off) = *(const v8bf*)(vB + (headOff + m0) * HD + off);
    }
#endif
    __syncthreads();

    v8f s[2];
#pragma unroll
    for (int t = 0; t < 2; ++t) {
      const int mbase = m0 + t * 16;
      // B-fragment of K^T from LDS: col = key row, K = head dim (contiguous 16)
      const __bf16* krow = kLds + (t * 16 + laneN) * HD + (half << 4);
      v8bf k0 = *(const v8bf*)krow;
      v8bf k1 = *(const v8bf*)(krow + 8);
      v16bf bK;
#pragma unroll
      for (int i = 0; i < 8; ++i) { bK[i] = k0[i]; bK[8 + i] = k1[i]; }
      v8f z = {};
      s[t] = __builtin_amdgcn_wmma_f32_16x16x32_bf16(false, aQ, false, bK,
                                                     (short)0, z, false, false);
#pragma unroll
      for (int r = 0; r < 8; ++r) {
        const int rowg = l0 + r + (half << 3);
        const int colg = mbase + laneN;
        float sv = s[t][r];
        if (rowg == colg) sv = sw[r];          // diag := self_w (pre-scale)
        sv *= scale;
        if (mask[((size_t)b * Ll + rowg) * Ll + colg] == 0) sv = -3.0e38f;
        s[t][r] = sv;
      }
    }

    // ---- online softmax update (row stats replicated across lane halves)
#pragma unroll
    for (int r = 0; r < 8; ++r) {
      const int rowg = l0 + r + (half << 3);
      float tmax = fmaxf(s[0][r], s[1][r]);
#pragma unroll
      for (int o = 1; o < 16; o <<= 1) tmax = fmaxf(tmax, __shfl_xor(tmax, o, 32));
      const float mnew = fmaxf(mrun[r], tmax);
      const float corr = __expf(mrun[r] - mnew);
      float p0 = __expf(s[0][r] - mnew);
      float p1 = __expf(s[1][r] - mnew);
      float rs = p0 + p1;
#pragma unroll
      for (int o = 1; o < 16; o <<= 1) rs += __shfl_xor(rs, o, 32);
      lrun[r] = lrun[r] * corr + rs;
      float dcon = 0.f;                        // diag prob -> self_v path
      if (rowg == m0 + laneN)      { dcon += p0; p0 = 0.f; }
      if (rowg == m0 + 16 + laneN) { dcon += p1; p1 = 0.f; }
#pragma unroll
      for (int o = 1; o < 16; o <<= 1) dcon += __shfl_xor(dcon, o, 32);
      pdiag[r] = pdiag[r] * corr + dcon;
      mrun[r] = mnew;
      acc0[r] *= corr;
      acc1[r] *= corr;
      const int rowl = r + (half << 3);
      pLds[rowl * 32 + laneN]      = f2bf(p0);
      pLds[rowl * 32 + 16 + laneN] = f2bf(p1);
    }
    __syncthreads();

    // ---- transpose P through LDS into an A-fragment, then y += P @ V
    v16bf aP;
    {
      const __bf16* prow = pLds + laneN * 32;
      v8bf p0 = *(const v8bf*)(prow + (half << 4));
      v8bf p1 = *(const v8bf*)(prow + (half << 4) + 8);
#pragma unroll
      for (int i = 0; i < 8; ++i) { aP[i] = p0[i]; aP[8 + i] = p1[i]; }
    }
    v16bf bV0, bV1;
#pragma unroll
    for (int i = 0; i < 16; ++i) {
      const __bf16* vrow = vLds + ((half << 4) + i) * HD;
      bV0[i] = vrow[laneN];
      bV1[i] = vrow[16 + laneN];
    }
    acc0 = __builtin_amdgcn_wmma_f32_16x16x32_bf16(false, aP, false, bV0,
                                                   (short)0, acc0, false, false);
    acc1 = __builtin_amdgcn_wmma_f32_16x16x32_bf16(false, aP, false, bV1,
                                                   (short)0, acc1, false, false);
    __syncthreads();
  }

  // ---- finalize: y = (acc + pdiag * self_v) / l -> bf16 [B*L, D]
#pragma unroll
  for (int r = 0; r < 8; ++r) {
    const int rowg = l0 + r + (half << 3);
    const float inv = 1.f / lrun[r];
    const __bf16* svrow = svB + (headOff + rowg) * HD;
    const float y0 = (acc0[r] + pdiag[r] * bf2f(svrow[laneN]))      * inv;
    const float y1 = (acc1[r] + pdiag[r] * bf2f(svrow[16 + laneN])) * inv;
    __bf16* yrow = yB + (size_t)(b * Ll + rowg) * Dd + h * HD;
    yrow[laneN]      = f2bf(y0);
    yrow[16 + laneN] = f2bf(y1);
  }
}

// ---------------------------------------------------------------------------
// Host-side orchestration
// ---------------------------------------------------------------------------
extern "C" void kernel_launch(void* const* d_in, const int* in_sizes, int n_in,
                              void* d_out, int out_size, void* d_ws, size_t ws_size,
                              hipStream_t stream) {
  (void)in_sizes; (void)n_in; (void)out_size; (void)ws_size;
  const float* obs = (const float*)d_in[0];
  const float* act = (const float*)d_in[1];
  const int*   msk = (const int*)d_in[2];
  const float* Wk  = (const float*)d_in[3];  const float* bk  = (const float*)d_in[4];
  const float* Wv  = (const float*)d_in[5];  const float* bv  = (const float*)d_in[6];
  const float* Wq  = (const float*)d_in[7];  const float* bq  = (const float*)d_in[8];
  const float* Wks = (const float*)d_in[9];  const float* bks = (const float*)d_in[10];
  const float* Wvs = (const float*)d_in[11]; const float* bvs = (const float*)d_in[12];
  const float* Wp  = (const float*)d_in[13]; const float* bp  = (const float*)d_in[14];

  char* ws = (char*)d_ws;
  const size_t nAct = (size_t)Mrows * Dd;
  __bf16* qb  = (__bf16*)ws; ws += nAct * sizeof(__bf16);
  __bf16* kb  = (__bf16*)ws; ws += nAct * sizeof(__bf16);
  __bf16* vb  = (__bf16*)ws; ws += nAct * sizeof(__bf16);
  __bf16* skb = (__bf16*)ws; ws += nAct * sizeof(__bf16);
  __bf16* svb = (__bf16*)ws; ws += nAct * sizeof(__bf16);
  __bf16* yb  = (__bf16*)ws; ws += nAct * sizeof(__bf16);
  float*  swb = (float*)ws;  ws += (size_t)Bb * Hh * Ll * sizeof(float);

  const dim3 gG(Mrows / 16, Dd / 32);
  gemm_bf16out_kernel<<<gG, 32, 0, stream>>>(obs, act, 2 * Dd, Wk,  bk,  kb);
  gemm_bf16out_kernel<<<gG, 32, 0, stream>>>(obs, act, 2 * Dd, Wv,  bv,  vb);
  gemm_bf16out_kernel<<<gG, 32, 0, stream>>>(obs, obs, Dd,     Wq,  bq,  qb);
  gemm_bf16out_kernel<<<gG, 32, 0, stream>>>(obs, obs, Dd,     Wks, bks, skb);
  gemm_bf16out_kernel<<<gG, 32, 0, stream>>>(obs, obs, Dd,     Wvs, bvs, svb);

  selfw_kernel<<<(Bb * Hh * Ll + 255) / 256, 256, 0, stream>>>(qb, skb, swb);

  attn_kernel<<<dim3(Ll / 16, Hh, Bb), 32, 0, stream>>>(qb, kb, vb, svb, swb,
                                                        msk, yb);

  gemm_f32out_kernel<<<gG, 32, 0, stream>>>(yb, Wp, bp, (float*)d_out);
}
